// VisionAttention_31722628448788
// MI455X (gfx1250) — matrix operand order repaired and checked by reading, not compiled, not run on hardware
//
#include <hip/hip_runtime.h>
#include <hip/hip_bf16.h>

typedef __attribute__((ext_vector_type(16))) _Float16 v16h;
typedef __attribute__((ext_vector_type(8)))  float    v8f;

#define SEQ   4096
#define HID   1280
#define NH    16
#define HD    80
#define HDP   96          // head dim padded to 3 chunks of 32
#define NQKV  3840
#define NSEQS 8

// ---------------------------------------------------------------------------
// WMMA wrapper
// ---------------------------------------------------------------------------
__device__ __forceinline__ v8f wmma_f16(v16h a, v16h b, v8f c) {
  return __builtin_amdgcn_wmma_f32_16x16x32_f16(false, a, false, b, (short)0, c,
                                                false, false);
}

// ---------------------------------------------------------------------------
// Async global -> LDS copy (CDNA5 GLOBAL_LOAD_ASYNC_TO_LDS_B128, ASYNCcnt).
// ldsaddr: low 32 bits of generic pointer = in-wave LDS byte offset.
// ---------------------------------------------------------------------------
__device__ __forceinline__ void async_g2l_b128(void* lds, const void* gptr) {
  unsigned ldsaddr = (unsigned)(unsigned long long)lds;
  unsigned long long ga = (unsigned long long)gptr;
  asm volatile("global_load_async_to_lds_b128 %0, %1, off"
               :: "v"(ldsaddr), "v"(ga) : "memory");
}
__device__ __forceinline__ void wait_async0() {
  asm volatile("s_wait_asynccnt 0x0" ::: "memory");
}

// ---------------------------------------------------------------------------
// Fragment loaders (ISA 7.12.2 layouts, wave32) — all loads are contiguous
// 16-byte runs per lane so they lower to b128 loads.
// ---------------------------------------------------------------------------
__device__ __forceinline__ v16h frag_a_contig(const _Float16* base, int ld, int lane) {
  int row  = lane & 15;
  int koff = (lane >> 4) << 3;
  const _Float16* p = base + row * ld;
  v16h a;
  *(uint4*)&a       = *(const uint4*)(p + koff);
  *((uint4*)&a + 1) = *(const uint4*)(p + 16 + koff);
  return a;
}

// B 32x16 f16 where storage is k-contiguous per column: element (k,n) at
// base[n*ld + k].  Lane col n = lane&15, upper half-wave carries k+8 (and k+24).
__device__ __forceinline__ v16h frag_bT_contig(const _Float16* base, int ld, int lane) {
  int n  = lane & 15;
  int kh = (lane >> 4) << 3;
  const _Float16* p = base + n * ld + kh;
  v16h b;
  *(uint4*)&b       = *(const uint4*)p;
  *((uint4*)&b + 1) = *(const uint4*)(p + 16);
  return b;
}

// ---------------------------------------------------------------------------
// f32 -> f16 convert (row-major passthrough)
// ---------------------------------------------------------------------------
__global__ void k_cvt_f16(const float* __restrict__ in, _Float16* __restrict__ out, int n) {
  int i = blockIdx.x * 256 + threadIdx.x;
  if (i < n) out[i] = (_Float16)in[i];
}

// ---------------------------------------------------------------------------
// f32 -> f16 convert + transpose: out[n][k] = in[k][n]   (LDS-tiled)
// ---------------------------------------------------------------------------
__global__ __launch_bounds__(256) void k_cvt_t(const float* __restrict__ in,
                                               _Float16* __restrict__ out,
                                               int K, int N) {
  __shared__ float tile[32][33];
  int kb = blockIdx.x * 32, nb = blockIdx.y * 32;
  int tx = threadIdx.x & 31, ty = threadIdx.x >> 5;
#pragma unroll
  for (int r = ty; r < 32; r += 8)
    tile[r][tx] = in[(size_t)(kb + r) * N + nb + tx];
  __syncthreads();
#pragma unroll
  for (int r = ty; r < 32; r += 8)
    out[(size_t)(nb + r) * K + kb + tx] = (_Float16)tile[tx][r];
}

// ---------------------------------------------------------------------------
// segment id / range per position
// ---------------------------------------------------------------------------
__global__ void k_seg(const int* __restrict__ cu, int* __restrict__ seg,
                      int* __restrict__ sstart, int* __restrict__ send) {
  int s = blockIdx.x * 256 + threadIdx.x;
  if (s >= SEQ) return;
  int g = 0;
#pragma unroll
  for (int j = 1; j <= NSEQS; ++j)
    if (s >= cu[j]) g = j;
  seg[s]    = g;
  sstart[s] = cu[g];
  send[s]   = cu[g + 1];
}

// ---------------------------------------------------------------------------
// Tiled f16 GEMM, f32 accumulate: C[MxN] = A[MxK] * BT^T + bias
// BT is the pre-transposed weight: BT[n][k].
// block = 128 threads = 4 waves; block tile 32x128; wave tile 16x64.
// Double-buffered async global->LDS staging: chunk i+1's DMA runs while
// chunk i's WMMAs execute; one barrier per iteration.
// ---------------------------------------------------------------------------
__global__ __launch_bounds__(128) void k_gemm(
    const _Float16* __restrict__ A, const _Float16* __restrict__ BT,
    const float* __restrict__ bias, float* __restrict__ C,
    int M, int N, int K) {
  __shared__ __align__(16) _Float16 sA [2][32][40];    // [buf][row][k]
  __shared__ __align__(16) _Float16 sBt[2][128][40];   // [buf][col][k]

  int t    = threadIdx.x;
  int wave = t >> 5;
  int lane = t & 31;
  int row0 = blockIdx.x * 32;
  int col0 = blockIdx.y * 128;
  int wrow = (wave >> 1) * 16;
  int wcol = (wave & 1) * 64;

  v8f acc[4];
#pragma unroll
  for (int i = 0; i < 4; ++i) acc[i] = (v8f){};

  int ar = t >> 2;           // A copy row: 0..31
  int ac = (t & 3) * 8;      // A copy k:   0,8,16,24

  auto stage = [&](int buf, int k0) {
    async_g2l_b128(&sA[buf][ar][ac], A + (size_t)(row0 + ar) * K + k0 + ac);
#pragma unroll
    for (int j = 0; j < 4; ++j) {
      int idx = t * 8 + j * 1024;
      int c   = idx >> 5;          // 0..127
      int kk  = idx & 31;          // 0,8,16,24
      async_g2l_b128(&sBt[buf][c][kk], BT + (size_t)(col0 + c) * K + k0 + kk);
    }
  };

  int nk = K >> 5;
  stage(0, 0);
  for (int i = 0; i < nk; ++i) {
    int cur = i & 1;
    wait_async0();           // chunk i resident in LDS (own wave's DMAs)
    __syncthreads();         // ... and everyone else's; prev reads of buf^1 done
    if (i + 1 < nk) {
      stage(cur ^ 1, (i + 1) << 5);                 // overlap DMA with WMMAs
      __builtin_prefetch(BT + (size_t)(col0 + t) * K + ((i + 2) << 5), 0, 1);
    }
    v16h af = frag_a_contig(&sA[cur][wrow][0], 40, lane);
    v16h bf[4];
#pragma unroll
    for (int tt = 0; tt < 4; ++tt)
      bf[tt] = frag_bT_contig(&sBt[cur][wcol + tt * 16][0], 40, lane);
#pragma unroll
    for (int tt = 0; tt < 4; ++tt)
      acc[tt] = wmma_f16(af, bf[tt], acc[tt]);
  }

  int halfS = lane >> 4;
  int colL  = lane & 15;
#pragma unroll
  for (int tt = 0; tt < 4; ++tt) {
    int nG   = col0 + wcol + tt * 16 + colL;
    float bv = bias[nG];
#pragma unroll
    for (int r = 0; r < 8; ++r) {
      int mG = row0 + wrow + 8 * halfS + r;
      C[(size_t)mG * N + nG] = acc[tt][r] + bv;
    }
  }
}

// ---------------------------------------------------------------------------
// RoPE + pack: q,k -> [NH][SEQ][HDP] f16 (zero-padded 80->96);
//              v  -> transposed [NH][HD][SEQ] f16 (key axis contiguous)
// ---------------------------------------------------------------------------
__global__ void k_rope_pack(const float* __restrict__ qkv,
                            const float* __restrict__ cosb,
                            const float* __restrict__ sinb,
                            _Float16* __restrict__ qp, _Float16* __restrict__ kp,
                            _Float16* __restrict__ vT) {
  int i = blockIdx.x * 256 + threadIdx.x;
  if (i >= SEQ * NH * HDP) return;
  int d = i % HDP;
  int h = (i / HDP) % NH;
  int s = i / (HDP * NH);
  size_t o = ((size_t)h * SEQ + s) * HDP + d;
  if (d >= HD) { qp[o] = (_Float16)0.f; kp[o] = (_Float16)0.f; return; }
  size_t rb = (size_t)s * NQKV;
  int col = h * HD + d;
  float qv = qkv[rb + col];
  float kv = qkv[rb + HID + col];
  float vv = qkv[rb + 2 * HID + col];
  float c  = cosb[s * HD + d];
  float sn = sinb[s * HD + d];
  int  d2  = (d < 40) ? d + 40 : d - 40;
  float sgn = (d < 40) ? -1.f : 1.f;
  float q2 = qkv[rb + h * HD + d2];
  float k2 = qkv[rb + HID + h * HD + d2];
  qp[o] = (_Float16)(qv * c + sgn * q2 * sn);
  kp[o] = (_Float16)(kv * c + sgn * k2 * sn);
  vT[((size_t)h * HD + d) * SEQ + s] = (_Float16)vv;
}

// ---------------------------------------------------------------------------
// Flash attention: one wave per (head, 16-query tile); varlen segment mask.
// Writes f16 [SEQ][HID] for the proj GEMM.
// ---------------------------------------------------------------------------
__global__ __launch_bounds__(128) void k_attn(
    const _Float16* __restrict__ qp, const _Float16* __restrict__ kp,
    const _Float16* __restrict__ vT, const int* __restrict__ seg,
    const int* __restrict__ sstart, const int* __restrict__ send,
    _Float16* __restrict__ attn) {
  __shared__ __align__(16) _Float16 sP[4][16][32];

  int wave = threadIdx.x >> 5;
  int lane = threadIdx.x & 31;
  int gw   = blockIdx.x * 4 + wave;     // 4096 waves = 16 heads * 256 qtiles
  int h    = gw >> 8;
  int q0   = (gw & 255) * 16;
  int halfS = lane >> 4;
  int colL  = lane & 15;
  const float scale = 0.11180339887498949f;   // 1/sqrt(80)

  // Q fragments for 3 K-chunks (head dim padded 80->96, pad is zero)
  const _Float16* qbase = qp + ((size_t)h * SEQ + q0) * HDP;
  v16h qa[3];
#pragma unroll
  for (int c = 0; c < 3; ++c) qa[c] = frag_a_contig(qbase + c * 32, HDP, lane);

  int segq[8];
#pragma unroll
  for (int r = 0; r < 8; ++r) segq[r] = seg[q0 + 8 * halfS + r];

  int kstart = sstart[q0];
  int kend   = send[q0 + 15];
  int kb0    = kstart & ~31;

  float m[8], l[8];
  v8f o[5];
#pragma unroll
  for (int r = 0; r < 8; ++r) { m[r] = -3.0e38f; l[r] = 0.f; }
#pragma unroll
  for (int tt = 0; tt < 5; ++tt) o[tt] = (v8f){};

  const _Float16* vhead = vT + (size_t)h * HD * SEQ;

  for (int kb = kb0; kb < kend; kb += 32) {
    // ---- scores: S = Q * K^T for two 16-key subtiles (batch frag loads)
    const _Float16* kb0p = kp + ((size_t)h * SEQ + kb) * HDP;
    const _Float16* kb1p = kp + ((size_t)h * SEQ + kb + 16) * HDP;
    v16h kf0[3], kf1[3];
#pragma unroll
    for (int c = 0; c < 3; ++c) {
      kf0[c] = frag_bT_contig(kb0p + c * 32, HDP, lane);
      kf1[c] = frag_bT_contig(kb1p + c * 32, HDP, lane);
    }
    v8f s0 = (v8f){}, s1 = (v8f){};
#pragma unroll
    for (int c = 0; c < 3; ++c) {
      s0 = wmma_f16(qa[c], kf0[c], s0);
      s1 = wmma_f16(qa[c], kf1[c], s1);
    }

    int sk0 = seg[kb + colL];
    int sk1 = seg[kb + 16 + colL];

    // ---- online softmax per row (row lives across 16 lanes of a half-wave)
#pragma unroll
    for (int r = 0; r < 8; ++r) {
      float v0 = s0[r] * scale, v1 = s1[r] * scale;
      bool ok0 = (segq[r] == sk0), ok1 = (segq[r] == sk1);
      v0 = ok0 ? v0 : -3.0e38f;
      v1 = ok1 ? v1 : -3.0e38f;
      float rm = fmaxf(v0, v1);
      rm = fmaxf(rm, __shfl_xor(rm, 1, 16));
      rm = fmaxf(rm, __shfl_xor(rm, 2, 16));
      rm = fmaxf(rm, __shfl_xor(rm, 4, 16));
      rm = fmaxf(rm, __shfl_xor(rm, 8, 16));
      float mn    = fmaxf(m[r], rm);
      float alpha = __expf(m[r] - mn);
      float p0 = ok0 ? __expf(v0 - mn) : 0.f;
      float p1 = ok1 ? __expf(v1 - mn) : 0.f;
      float rs = p0 + p1;
      rs += __shfl_xor(rs, 1, 16);
      rs += __shfl_xor(rs, 2, 16);
      rs += __shfl_xor(rs, 4, 16);
      rs += __shfl_xor(rs, 8, 16);
      l[r] = l[r] * alpha + rs;
      m[r] = mn;
#pragma unroll
      for (int tt = 0; tt < 5; ++tt) o[tt][r] *= alpha;
      sP[wave][8 * halfS + r][colL]      = (_Float16)p0;
      sP[wave][8 * halfS + r][16 + colL] = (_Float16)p1;
    }
    // LDS round-trip: C-layout -> A-layout (same wave only; wait ds writes)
    asm volatile("s_wait_dscnt 0x0" ::: "memory");
    v16h pa = frag_a_contig(&sP[wave][0][0], 32, lane);

    // ---- O += P * V  (N = 80 -> 5 tiles); vT[h][d][s] is key-contiguous
    v16h vf[5];
#pragma unroll
    for (int tt = 0; tt < 5; ++tt)
      vf[tt] = frag_bT_contig(vhead + (size_t)(tt * 16) * SEQ + kb, SEQ, lane);
#pragma unroll
    for (int tt = 0; tt < 5; ++tt)
      o[tt] = wmma_f16(pa, vf[tt], o[tt]);
  }

  // ---- normalize and write f16 output [SEQ][HID]
#pragma unroll
  for (int r = 0; r < 8; ++r) {
    float inv = (l[r] > 0.f) ? (1.f / l[r]) : 0.f;
    int mG = q0 + 8 * halfS + r;
#pragma unroll
    for (int tt = 0; tt < 5; ++tt)
      attn[(size_t)mG * HID + h * HD + tt * 16 + colL] = (_Float16)(o[tt][r] * inv);
  }
}

// ---------------------------------------------------------------------------
// host-side orchestration
// ---------------------------------------------------------------------------
extern "C" void kernel_launch(void* const* d_in, const int* in_sizes, int n_in,
                              void* d_out, int out_size, void* d_ws, size_t ws_size,
                              hipStream_t stream) {
  const float* x      = (const float*)d_in[0];
  const int*   cu     = (const int*)  d_in[1];
  const float* cosb   = (const float*)d_in[2];
  const float* sinb   = (const float*)d_in[3];
  const float* w_qkv  = (const float*)d_in[4];
  const float* b_qkv  = (const float*)d_in[5];
  const float* w_proj = (const float*)d_in[6];
  const float* b_proj = (const float*)d_in[7];
  float* out = (float*)d_out;

  char* ws = (char*)d_ws;
  size_t off = 0;
  auto nxt = [&](size_t b) { char* p = ws + off; off = (off + b + 255) & ~(size_t)255; return p; };

  _Float16* xh   = (_Float16*)nxt((size_t)SEQ * HID * 2);
  _Float16* wqT  = (_Float16*)nxt((size_t)HID * NQKV * 2);   // [NQKV][HID]
  _Float16* wpT  = (_Float16*)nxt((size_t)HID * HID * 2);    // [HID][HID]
  float*    qkv  = (float*)   nxt((size_t)SEQ * NQKV * 4);
  _Float16* qpk  = (_Float16*)nxt((size_t)NH * SEQ * HDP * 2);
  _Float16* kpk  = (_Float16*)nxt((size_t)NH * SEQ * HDP * 2);
  _Float16* vTp  = (_Float16*)nxt((size_t)NH * HD * SEQ * 2);
  _Float16* atn  = (_Float16*)nxt((size_t)SEQ * HID * 2);
  int* seg    = (int*)nxt(SEQ * 4);
  int* sstart = (int*)nxt(SEQ * 4);
  int* send   = (int*)nxt(SEQ * 4);

  // 1) operand conversion: x passthrough, weights transposed
  int nx = SEQ * HID;
  k_cvt_f16<<<(nx + 255) / 256, 256, 0, stream>>>(x, xh, nx);
  k_cvt_t<<<dim3(HID / 32, NQKV / 32), 256, 0, stream>>>(w_qkv,  wqT, HID, NQKV);
  k_cvt_t<<<dim3(HID / 32, HID  / 32), 256, 0, stream>>>(w_proj, wpT, HID, HID);

  // 2) segment table
  k_seg<<<(SEQ + 255) / 256, 256, 0, stream>>>(cu, seg, sstart, send);

  // 3) QKV GEMM: (4096x1280)*(1280x3840)+b
  k_gemm<<<dim3(SEQ / 32, NQKV / 128), 128, 0, stream>>>(xh, wqT, b_qkv, qkv,
                                                         SEQ, NQKV, HID);

  // 4) RoPE + pack (q,k padded row-major; v transposed)
  int npack = SEQ * NH * HDP;
  k_rope_pack<<<(npack + 255) / 256, 256, 0, stream>>>(qkv, cosb, sinb, qpk, kpk, vTp);

  // 5) varlen flash attention (1 wave per head x 16-query tile)
  k_attn<<<(NH * (SEQ / 16)) / 4, 128, 0, stream>>>(qpk, kpk, vTp, seg, sstart, send, atn);

  // 6) proj GEMM: (4096x1280)*(1280x1280)+b -> d_out (f32)
  k_gemm<<<dim3(SEQ / 32, HID / 128), 128, 0, stream>>>(atn, wpT, b_proj, out,
                                                        SEQ, HID, HID);
}
